// GroupedQueryAttention_3032246911204
// MI455X (gfx1250) — compile-verified, matrix operand order checked by reading, not run on hardware
//
#include <hip/hip_runtime.h>
#include <hip/hip_bf16.h>

// ---------------------------------------------------------------------------
// GQA attention layer for MI455X (gfx1250), bf16 WMMA pipeline.
//   B=2 S=2048 H=32 KV=8 D=128 E=4096 KVE=1024
// Pipeline: f32->bf16 converts -> 3 WMMA GEMMs (Q,K,V) -> RoPE -> flash
// attention (WMMA QK^T + online softmax + WMMA PV) -> WMMA GEMM (f32 out).
// Non-transposed LDS tile staging uses gfx1250 async copies
// (global_load_async_to_lds_b128 + s_wait_asynccnt).
// ---------------------------------------------------------------------------

typedef __attribute__((ext_vector_type(16))) __bf16 v16bf;
typedef __attribute__((ext_vector_type(8)))  float  v8f;

static __device__ __forceinline__ unsigned short f2bf(float f) {
    unsigned u = __float_as_uint(f);
    return (unsigned short)((u + 0x7FFFu + ((u >> 16) & 1u)) >> 16);
}
static __device__ __forceinline__ float bf2f(unsigned short h) {
    return __uint_as_float(((unsigned)h) << 16);
}
static __device__ __forceinline__ v8f v8f_zero() {
    v8f z = {0.f, 0.f, 0.f, 0.f, 0.f, 0.f, 0.f, 0.f};
    return z;
}

// LDS byte address (32-bit, addrspace(3) offset) of a shared-memory pointer.
static __device__ __forceinline__ unsigned lds_addr_of(const void* p) {
    return (unsigned)(unsigned long long)
        (const __attribute__((address_space(3))) void*)p;
}

// Async 16-byte global -> LDS copy (gfx1250 ASYNCcnt path).
static __device__ __forceinline__ void async_copy16(unsigned lds_addr,
                                                    const void* gptr) {
    asm volatile("global_load_async_to_lds_b128 %0, %1, off"
                 :
                 : "v"(lds_addr), "v"((unsigned long long)gptr)
                 : "memory");
}
static __device__ __forceinline__ void wait_async0() {
    asm volatile("s_wait_asynccnt 0x0" ::: "memory");
}

// ---------------------------------------------------------------------------
// fp32 -> bf16 (RNE) elementwise
// ---------------------------------------------------------------------------
__global__ void cvt_f32_bf16(const float* __restrict__ in,
                             unsigned short* __restrict__ out, int n) {
    int i = blockIdx.x * blockDim.x + threadIdx.x;
    if (i < n) out[i] = f2bf(in[i]);
}

// ---------------------------------------------------------------------------
// bf16 GEMM: C[M,N] = A[M,K] * B[K,N], f32 accumulate via v_wmma_f32_16x16x32_bf16
// Block: 256 threads (8 waves), BM=BN=128, BK=32. Wave grid 4(M) x 2(N),
// each wave computes a 32x64 tile = 2x4 WMMA accumulators.
// A tile staged with async global->LDS copies; B tile transposed via VGPRs.
// ---------------------------------------------------------------------------
template <bool OUT_F32>
__global__ __launch_bounds__(256)
void gemm_bf16_wmma(const unsigned short* __restrict__ A,
                    const unsigned short* __restrict__ B,
                    void* __restrict__ C, int M, int N, int K) {
    constexpr int BM = 128, BN = 128, BK = 32;
    constexpr int LDT = BK + 8;  // 40 shorts = 80B rows: 16B aligned
    __shared__ unsigned short sA[BM * LDT];  // [m][k]
    __shared__ unsigned short sB[BN * LDT];  // [n][k] (transposed)

    const int tid  = threadIdx.x;
    const int lane = tid & 31;
    const int wave = tid >> 5;
    const int waveM = wave >> 1;  // 0..3
    const int waveN = wave & 1;   // 0..1
    const int l16 = lane & 15;
    const int lhi = lane >> 4;
    const int bm0 = blockIdx.x * BM;
    const int bn0 = blockIdx.y * BN;
    const unsigned sA_base = lds_addr_of(sA);

    v8f acc[2][4];
    for (int i = 0; i < 2; ++i)
        for (int j = 0; j < 4; ++j) acc[i][j] = v8f_zero();

    for (int kt = 0; kt < K; kt += BK) {
        // A tile 128x32: async global->LDS, 256 threads * 2 * 16B
        for (int i = 0; i < 2; ++i) {
            int e = (tid * 2 + i) * 8;
            int r = e >> 5, c = e & 31;
            async_copy16(sA_base + (unsigned)(r * LDT + c) * 2u,
                         A + (size_t)(bm0 + r) * K + kt + c);
        }
        // B tile 32x128, scattered transpose into sB[n][k]
        for (int i = 0; i < 2; ++i) {
            int e = (tid * 2 + i) * 8;
            int kk = e >> 7, n = e & 127;
            uint4 v = *(const uint4*)(B + (size_t)(kt + kk) * N + bn0 + n);
            const unsigned short* pv = (const unsigned short*)&v;
#pragma unroll
            for (int j = 0; j < 8; ++j) sB[(n + j) * LDT + kk] = pv[j];
        }
        // prefetch next k-tile while this one is consumed
        if (kt + BK < K) {
            int e = tid * 16;
            int ra = e >> 5;
            int rb = e >> 7;
            __builtin_prefetch(A + (size_t)(bm0 + ra) * K + kt + BK, 0, 0);
            __builtin_prefetch(B + (size_t)(kt + BK + rb) * N + bn0 + (tid & 7) * 16,
                               0, 0);
        }
        wait_async0();
        __syncthreads();

        v16bf afrag[2], bfrag[4];
#pragma unroll
        for (int mi = 0; mi < 2; ++mi) {
            int row = waveM * 32 + mi * 16 + l16;
            afrag[mi] = *(const v16bf*)(sA + row * LDT + lhi * 16);
        }
#pragma unroll
        for (int ni = 0; ni < 4; ++ni) {
            int col = waveN * 64 + ni * 16 + l16;
            bfrag[ni] = *(const v16bf*)(sB + col * LDT + lhi * 16);
        }
#pragma unroll
        for (int mi = 0; mi < 2; ++mi)
#pragma unroll
            for (int ni = 0; ni < 4; ++ni)
                acc[mi][ni] = __builtin_amdgcn_wmma_f32_16x16x32_bf16(
                    false, afrag[mi], false, bfrag[ni], (short)0, acc[mi][ni],
                    false, false);
        __syncthreads();
    }

    // C/D layout: vgpr r, lanes 0-15 -> row r, lanes 16-31 -> row r+8
#pragma unroll
    for (int mi = 0; mi < 2; ++mi)
#pragma unroll
        for (int ni = 0; ni < 4; ++ni)
#pragma unroll
            for (int r = 0; r < 8; ++r) {
                int row = bm0 + waveM * 32 + mi * 16 + r + 8 * lhi;
                int col = bn0 + waveN * 64 + ni * 16 + l16;
                float v = acc[mi][ni][r];
                if (OUT_F32)
                    ((float*)C)[(size_t)row * N + col] = v;
                else
                    ((unsigned short*)C)[(size_t)row * N + col] = f2bf(v);
            }
}

// ---------------------------------------------------------------------------
// RoPE (rotate-half), in place on bf16 [BS, H*D], pos = row % S
// ---------------------------------------------------------------------------
__global__ void rope_kernel(unsigned short* __restrict__ q, int BS, int H,
                            int D, int S) {
    int half = D >> 1;
    int idx = blockIdx.x * blockDim.x + threadIdx.x;
    int total = BS * H * half;
    if (idx >= total) return;
    int d = idx % half;
    int t = idx / half;
    int h = t % H;
    int row = t / H;
    int pos = row % S;
    size_t base = (size_t)row * H * D + (size_t)h * D;
    float x0 = bf2f(q[base + d]);
    float x1 = bf2f(q[base + d + half]);
    float ang = (float)pos * __powf(10000.0f, -2.0f * (float)d / (float)D);
    float s, c;
    __sincosf(ang, &s, &c);
    q[base + d]        = f2bf(x0 * c - x1 * s);
    q[base + d + half] = f2bf(x1 * c + x0 * s);
}

// ---------------------------------------------------------------------------
// Flash attention (causal, GQA). Grid: (B*H, S/128). Block: 256 thr = 8 waves.
// Each wave owns 16 query rows; K/V tiles of 32 positions staged in LDS and
// shared by all waves. K tile staged via async global->LDS (no transpose
// needed for the QK^T B-fragment); V transposed into LDS for PV; P
// round-trips through per-wave LDS to form the A-fragment.
// ---------------------------------------------------------------------------
__global__ __launch_bounds__(256)
void flash_attn(const unsigned short* __restrict__ Q,   // [B*S, H*D] (roped)
                const unsigned short* __restrict__ Kc,  // [B*S, KV*D] (roped)
                const unsigned short* __restrict__ V,   // [B*S, KV*D]
                unsigned short* __restrict__ O,         // [B*S, H*D]
                int B_, int S, int H, int KV, int D) {
    constexpr int LDK = 128 + 8;  // 136 shorts: 272B rows (16B aligned)
    constexpr int LDV = 32 + 8;   // 40 shorts
    constexpr int LDP = 32 + 8;   // 40 shorts
    __shared__ unsigned short sK[32 * LDK];       // [kvpos][dim]
    __shared__ unsigned short sV[128 * LDV];      // [dim][kvpos]
    __shared__ unsigned short sP[8 * 16 * LDP];   // per-wave P tile

    const int tid = threadIdx.x, lane = tid & 31, wave = tid >> 5;
    const int l16 = lane & 15, lhi = lane >> 4;
    const int bh = blockIdx.x;
    const int b = bh / H, h = bh % H;
    const int hkv = h / (H / KV);
    const int qb = blockIdx.y;
    const int q0 = qb * 128 + wave * 16;
    const int E = H * D, KVE = KV * D;
    const float scale = 0.08838834764831845f;  // 1/sqrt(128)
    const unsigned sK_base = lds_addr_of(sK);

    // preload this wave's Q fragments (16 rows x 128 dims -> 4 k-chunks)
    v16bf qfrag[4];
    {
        const unsigned short* qp =
            Q + (size_t)(b * S + q0 + l16) * E + h * D + lhi * 16;
#pragma unroll
        for (int kk = 0; kk < 4; ++kk)
            qfrag[kk] = *(const v16bf*)(qp + kk * 32);
    }

    v8f o[8];
#pragma unroll
    for (int i = 0; i < 8; ++i) o[i] = v8f_zero();
    float mrow[8], lrow[8];
#pragma unroll
    for (int r = 0; r < 8; ++r) { mrow[r] = -1e30f; lrow[r] = 0.f; }

    const int nTiles = (qb + 1) * 4;  // causal: kv positions up to q-block end
    for (int t = 0; t < nTiles; ++t) {
        __syncthreads();  // protect previous tile's LDS until PV done
        const int p0 = t * 32;
        // K tile 32x128 -> sK row-major via async copies
        for (int i = 0; i < 2; ++i) {
            int e = (tid * 2 + i) * 8;
            int r = e >> 7, d = e & 127;
            async_copy16(sK_base + (unsigned)(r * LDK + d) * 2u,
                         Kc + (size_t)(b * S + p0 + r) * KVE + hkv * D + d);
        }
        // V tile 32x128 -> sV transposed [dim][kvpos]
        for (int i = 0; i < 2; ++i) {
            int e = (tid * 2 + i) * 8;
            int r = e >> 7, d = e & 127;
            uint4 v = *(const uint4*)(V + (size_t)(b * S + p0 + r) * KVE +
                                      hkv * D + d);
            const unsigned short* pv = (const unsigned short*)&v;
#pragma unroll
            for (int j = 0; j < 8; ++j) sV[(d + j) * LDV + r] = pv[j];
        }
        // prefetch next tile's K/V
        if (t + 1 < nTiles) {
            int e = tid * 16;
            int r = e >> 7, d = e & 127;
            __builtin_prefetch(
                Kc + (size_t)(b * S + p0 + 32 + r) * KVE + hkv * D + d, 0, 0);
            __builtin_prefetch(
                V + (size_t)(b * S + p0 + 32 + r) * KVE + hkv * D + d, 0, 0);
        }
        wait_async0();
        __syncthreads();

        // scores S = Q K^T : two 16x16 tiles over this 32-wide kv tile
        v8f sc[2];
#pragma unroll
        for (int ni = 0; ni < 2; ++ni) {
            v8f c = v8f_zero();
#pragma unroll
            for (int kk = 0; kk < 4; ++kk) {
                v16bf bfr = *(const v16bf*)(sK + (ni * 16 + l16) * LDK +
                                            kk * 32 + lhi * 16);
                c = __builtin_amdgcn_wmma_f32_16x16x32_bf16(
                    false, qfrag[kk], false, bfr, (short)0, c, false, false);
            }
            sc[ni] = c;
        }
        // scale + causal mask (first tile always has a valid column per row)
#pragma unroll
        for (int ni = 0; ni < 2; ++ni)
#pragma unroll
            for (int r = 0; r < 8; ++r) {
                int row = q0 + r + 8 * lhi;
                int col = p0 + ni * 16 + l16;
                float v = sc[ni][r] * scale;
                sc[ni][r] = (col <= row) ? v : -1e30f;
            }

        // online softmax: row max/sum via 16-lane xor-shuffle reductions
        float newm[8];
        float alpha[8];
#pragma unroll
        for (int r = 0; r < 8; ++r) {
            float v = fmaxf(sc[0][r], sc[1][r]);
            for (int off = 1; off < 16; off <<= 1)
                v = fmaxf(v, __shfl_xor(v, off, 32));
            newm[r] = fmaxf(mrow[r], v);
            alpha[r] = __expf(mrow[r] - newm[r]);
            mrow[r] = newm[r];
        }
        unsigned short* pw = sP + wave * 16 * LDP;
#pragma unroll
        for (int r = 0; r < 8; ++r) {
            float p0f = __expf(sc[0][r] - newm[r]);
            float p1f = __expf(sc[1][r] - newm[r]);
            float sum = p0f + p1f;
            for (int off = 1; off < 16; off <<= 1)
                sum += __shfl_xor(sum, off, 32);
            lrow[r] = lrow[r] * alpha[r] + sum;
            int rl = r + 8 * lhi;
            pw[rl * LDP + l16]      = f2bf(p0f);
            pw[rl * LDP + 16 + l16] = f2bf(p1f);
#pragma unroll
            for (int dt = 0; dt < 8; ++dt) o[dt][r] *= alpha[r];
        }
        __syncthreads();  // P visible; K/V stable until next loop-top barrier

        // O += P @ V  (A = P 16x32 from LDS, B = V^T columns from sV)
        v16bf pa = *(const v16bf*)(pw + l16 * LDP + lhi * 16);
#pragma unroll
        for (int dt = 0; dt < 8; ++dt) {
            v16bf vb = *(const v16bf*)(sV + (dt * 16 + l16) * LDV + lhi * 16);
            o[dt] = __builtin_amdgcn_wmma_f32_16x16x32_bf16(
                false, pa, false, vb, (short)0, o[dt], false, false);
        }
    }

    // normalize and store bf16 attention output [B*S, H*D]
#pragma unroll
    for (int r = 0; r < 8; ++r) {
        float inv = 1.0f / lrow[r];
        int row = q0 + r + 8 * lhi;
        size_t base = (size_t)(b * S + row) * E + (size_t)h * D;
#pragma unroll
        for (int dt = 0; dt < 8; ++dt)
            O[base + dt * 16 + l16] = f2bf(o[dt][r] * inv);
    }
}

// ---------------------------------------------------------------------------
// Host launch
// ---------------------------------------------------------------------------
extern "C" void kernel_launch(void* const* d_in, const int* in_sizes, int n_in,
                              void* d_out, int out_size, void* d_ws,
                              size_t ws_size, hipStream_t stream) {
    (void)in_sizes; (void)n_in; (void)out_size; (void)ws_size;
    const int B = 2, S = 2048, H = 32, KV = 8, D = 128;
    const int E = H * D, KVE = KV * D, BS = B * S;

    const float* x  = (const float*)d_in[0];
    const float* Wq = (const float*)d_in[1];
    const float* Wk = (const float*)d_in[2];
    const float* Wv = (const float*)d_in[3];
    const float* Wo = (const float*)d_in[4];
    float* out = (float*)d_out;

    char* ws = (char*)d_ws;
    size_t off = 0;
    auto take = [&](size_t elems) {
        unsigned short* p = (unsigned short*)(ws + off);
        off += elems * sizeof(unsigned short);
        return p;
    };
    unsigned short* xb  = take((size_t)BS * E);
    unsigned short* Wqb = take((size_t)E * E);
    unsigned short* Wkb = take((size_t)E * KVE);
    unsigned short* Wvb = take((size_t)E * KVE);
    unsigned short* Wob = take((size_t)E * E);
    unsigned short* Qb  = take((size_t)BS * E);
    unsigned short* Kb  = take((size_t)BS * KVE);
    unsigned short* Vb  = take((size_t)BS * KVE);
    unsigned short* Ob  = xb;  // x no longer needed after QKV projections

    const int thr = 256;
    auto blocks = [&](size_t n) { return (unsigned)((n + thr - 1) / thr); };

    cvt_f32_bf16<<<blocks((size_t)BS * E), thr, 0, stream>>>(x, xb, BS * E);
    cvt_f32_bf16<<<blocks((size_t)E * E), thr, 0, stream>>>(Wq, Wqb, E * E);
    cvt_f32_bf16<<<blocks((size_t)E * KVE), thr, 0, stream>>>(Wk, Wkb, E * KVE);
    cvt_f32_bf16<<<blocks((size_t)E * KVE), thr, 0, stream>>>(Wv, Wvb, E * KVE);
    cvt_f32_bf16<<<blocks((size_t)E * E), thr, 0, stream>>>(Wo, Wob, E * E);

    gemm_bf16_wmma<false><<<dim3(BS / 128, E / 128), 256, 0, stream>>>(
        xb, Wqb, Qb, BS, E, E);
    gemm_bf16_wmma<false><<<dim3(BS / 128, KVE / 128), 256, 0, stream>>>(
        xb, Wkb, Kb, BS, KVE, E);
    gemm_bf16_wmma<false><<<dim3(BS / 128, KVE / 128), 256, 0, stream>>>(
        xb, Wvb, Vb, BS, KVE, E);

    rope_kernel<<<blocks((size_t)BS * H * (D / 2)), thr, 0, stream>>>(
        Qb, BS, H, D, S);
    rope_kernel<<<blocks((size_t)BS * KV * (D / 2)), thr, 0, stream>>>(
        Kb, BS, KV, D, S);

    flash_attn<<<dim3(B * H, S / 128), 256, 0, stream>>>(Qb, Kb, Vb, Ob, B, S,
                                                         H, KV, D);

    gemm_bf16_wmma<true><<<dim3(BS / 128, E / 128), 256, 0, stream>>>(
        Ob, Wob, out, BS, E, E);
}
